// ProtoNet_45646912422244
// MI455X (gfx1250) — matrix operand order repaired and checked by reading, not compiled, not run on hardware
//
#include <hip/hip_runtime.h>

// ---------------------------------------------------------------------------
// ProtoNet head on MI455X (gfx1250):
//   C[640x1600] = [supp;query][640x21168] @ W_enc[21168x1600]   (bf16 WMMA, f32 acc)
//   proto = mean_k(C[:160]), row norms, cosine sim, log_softmax
//
// Roofline: ~193 MB compulsory HBM traffic -> ~8.3 us floor at 23.3 TB/s.
// A (54 MB) and W_enc (135 MB) fit in the 192 MB L2, so cross-tile re-reads
// are L2 hits and the GEMM is effectively single-pass from HBM. bf16 WMMA
// (16x16x32) keeps the 43.4 GFLOP compute at the same order as that floor;
// the f32 WMMA form (16x16x4) would be ~8x slower per instruction.
// ---------------------------------------------------------------------------

typedef __attribute__((ext_vector_type(16))) __bf16 v16bf;
typedef __attribute__((ext_vector_type(8)))  __bf16 v8bf;
typedef __attribute__((ext_vector_type(8)))  float  v8f;

#define FEAT   21168          // 3*84*84
#define DIM    1600
#define NWAY   32
#define KSHOT  5
#define QN     15
#define MSUP   160            // 32*5 support rows
#define MQRY   480            // 32*15 query rows
#define MROWS  640            // MSUP + MQRY

#define BM 128                // block tile M (4 waves x 32 rows)
#define BN 64                 // block tile N
#define BK 64                 // staged K per barrier (two WMMA K-halves)
#define BLK 128               // threads per block (4 waves)
#define KT     ((FEAT + BK - 1) / BK)       // 331 steps; last has only 48 valid k
#define SPLITK 8
#define KCHUNK ((KT + SPLITK - 1) / SPLITK) // 42

union U16bf { v16bf v; v8bf h[2]; };

static __device__ __forceinline__ v8f wmma_bf16(v16bf a, v16bf b, v8f c) {
    return __builtin_amdgcn_wmma_f32_16x16x32_bf16(false, a, false, b,
                                                   (short)0, c, false, false);
}

// One staged K-step: fill LDS (fp32 -> bf16), sync, 16 WMMA per wave, sync.
// TAIL=true compiles the chunk-granular zero-fill predicates; the main loop
// instantiates TAIL=false and is completely branch-free.
template <bool TAIL>
static __device__ __forceinline__
void gemm_step(int k0, const float* __restrict__ arow_ptr,
               const float* __restrict__ Wenc, int tid, int lane, int wave,
               int n0, __bf16 (*lA)[BK], __bf16 (*lB)[BK], v8f acc[2][4])
{
    // ---- stage A: thread 'tid' converts one full 64-float row segment ----
    {
        const float* src = arow_ptr + k0;
        #pragma unroll
        for (int c = 0; c < BK / 16; ++c) {            // 16-float chunks
            float t[16];
            if (!TAIL || (k0 + c * 16 < FEAT)) {       // FEAT is 16-aligned
                const float4* p = (const float4*)(src + c * 16);
                float4 f0 = p[0], f1 = p[1], f2 = p[2], f3 = p[3];
                t[0]=f0.x;  t[1]=f0.y;  t[2]=f0.z;  t[3]=f0.w;
                t[4]=f1.x;  t[5]=f1.y;  t[6]=f1.z;  t[7]=f1.w;
                t[8]=f2.x;  t[9]=f2.y;  t[10]=f2.z; t[11]=f2.w;
                t[12]=f3.x; t[13]=f3.y; t[14]=f3.z; t[15]=f3.w;
            } else {
                #pragma unroll
                for (int i = 0; i < 16; ++i) t[i] = 0.0f;
            }
            v8bf o0, o1;
            #pragma unroll
            for (int i = 0; i < 8; ++i) { o0[i] = (__bf16)t[i]; o1[i] = (__bf16)t[8 + i]; }
            *(v8bf*)&lA[tid][c * 16]     = o0;
            *(v8bf*)&lA[tid][c * 16 + 8] = o1;
        }
        if (!TAIL) __builtin_prefetch(src + BK, 0, 1); // speculative, next tile
    }

    // ---- stage B (transposed in LDS): column n, 32 k values per thread ----
    {
        const int bn  = tid & 63;
        const int bk0 = (tid >> 6) * 32;               // 0 or 32
        const float* bp = Wenc + (size_t)(k0 + bk0) * DIM + n0 + bn;
        #pragma unroll
        for (int c = 0; c < 4; ++c) {                  // 8-k chunks
            float t[8];
            if (!TAIL || (k0 + bk0 + c * 8 < FEAT)) {  // FEAT is 8-aligned
                #pragma unroll
                for (int i = 0; i < 8; ++i) t[i] = bp[(size_t)(c * 8 + i) * DIM];
            } else {
                #pragma unroll
                for (int i = 0; i < 8; ++i) t[i] = 0.0f;
            }
            v8bf ob;
            #pragma unroll
            for (int i = 0; i < 8; ++i) ob[i] = (__bf16)t[i];
            *(v8bf*)&lB[bn][bk0 + c * 8] = ob;
        }
        if (!TAIL) __builtin_prefetch(bp + (size_t)BK * DIM, 0, 1);
    }

    __syncthreads();

    // ---- 2 (M) x 4 (N) x 2 (K-half) = 16 WMMA per wave ----
    const int kl = lane >> 4;                          // K-half selector (contiguous k order)
    const int lr = lane & 15;
    const int wm = wave * 32;
    #pragma unroll
    for (int kh = 0; kh < 2; ++kh) {
        const int ko = kh * 32;
        U16bf a[2];
        #pragma unroll
        for (int mi = 0; mi < 2; ++mi) {
            // A 16x32: elems 0..15 = K(kl*16+0..15)
            a[mi].v = *(const v16bf*)&lA[wm + mi * 16 + lr][ko + kl * 16];
        }
        #pragma unroll
        for (int ni = 0; ni < 4; ++ni) {
            // B 32x16: fixed column, elems 0..15 = K(kl*16+0..15) -> one 32B load
            v16bf b = *(const v16bf*)&lB[ni * 16 + lr][ko + kl * 16];
            acc[0][ni] = wmma_bf16(a[0].v, b, acc[0][ni]);
            acc[1][ni] = wmma_bf16(a[1].v, b, acc[1][ni]);
        }
    }

    __syncthreads();
}

// ---------------------------------------------------------------------------
// Kernel 1: split-K tiled GEMM, fp32 inputs converted to bf16 in LDS,
//           v_wmma_f32_16x16x32_bf16 accumulation, atomic f32 reduction.
// ---------------------------------------------------------------------------
__global__ __launch_bounds__(BLK)
void gemm_bf16_wmma(const float* __restrict__ supp,
                    const float* __restrict__ query,
                    const float* __restrict__ Wenc,
                    float* __restrict__ C)
{
    __shared__ __align__(128) __bf16 lA[BM][BK];   // 16 KB
    __shared__ __align__(128) __bf16 lB[BN][BK];   //  8 KB, transposed (k contiguous)

    const int tid  = threadIdx.x;
    const int lane = tid & 31;
    const int wave = tid >> 5;

    const int m0 = blockIdx.y * BM;
    const int n0 = blockIdx.x * BN;

    const int kt_beg = blockIdx.z * KCHUNK;
    const int kt_end = (kt_beg + KCHUNK < KT) ? (kt_beg + KCHUNK) : KT;
    const bool has_tail    = (kt_end == KT);       // only last split owns ragged tile
    const int  kt_full_end = kt_end - (has_tail ? 1 : 0);

    const int gm = m0 + tid;                       // staging row for this thread
    const float* arow_ptr = (gm < MSUP) ? (supp  + (size_t)gm * FEAT)
                                        : (query + (size_t)(gm - MSUP) * FEAT);

    v8f acc[2][4] = {};

    for (int kt = kt_beg; kt < kt_full_end; ++kt)
        gemm_step<false>(kt * BK, arow_ptr, Wenc, tid, lane, wave, n0, lA, lB, acc);
    if (has_tail)
        gemm_step<true>((KT - 1) * BK, arow_ptr, Wenc, tid, lane, wave, n0, lA, lB, acc);

    // ---- epilogue: C/D layout -> lane holds col (lane&15), rows v + 8*(lane>>4) ----
    const int kl = lane >> 4, lr = lane & 15;
    #pragma unroll
    for (int mi = 0; mi < 2; ++mi) {
        #pragma unroll
        for (int ni = 0; ni < 4; ++ni) {
            const int row0 = m0 + wave * 32 + mi * 16 + kl * 8;
            const int col  = n0 + ni * 16 + lr;
            #pragma unroll
            for (int v = 0; v < 8; ++v)
                atomicAdd(&C[(size_t)(row0 + v) * DIM + col], acc[mi][ni][v]);
        }
    }
}

// ---------------------------------------------------------------------------
// Kernel 2: class prototypes = mean over 5 support rows
// ---------------------------------------------------------------------------
__global__ void proto_mean(const float* __restrict__ C, float* __restrict__ proto)
{
    int idx = blockIdx.x * blockDim.x + threadIdx.x;
    if (idx >= NWAY * DIM) return;
    int n = idx / DIM, d = idx - n * DIM;
    const float* base = C + (size_t)(n * KSHOT) * DIM + d;
    float s = 0.0f;
    #pragma unroll
    for (int k = 0; k < KSHOT; ++k) s += base[(size_t)k * DIM];
    proto[idx] = s * (1.0f / KSHOT);
}

// ---------------------------------------------------------------------------
// Kernel 3: inverse L2 norms (eps-clamped) of prototype and query rows
// ---------------------------------------------------------------------------
__global__ void row_norms(const float* __restrict__ C, const float* __restrict__ proto,
                          float* __restrict__ pinv, float* __restrict__ qinv)
{
    const int b = blockIdx.x;
    const float* row = (b < NWAY) ? (proto + (size_t)b * DIM)
                                  : (C + (size_t)(MSUP + (b - NWAY)) * DIM);
    float s = 0.0f;
    for (int d = threadIdx.x; d < DIM; d += blockDim.x) { float v = row[d]; s += v * v; }
    #pragma unroll
    for (int off = 16; off >= 1; off >>= 1) s += __shfl_xor(s, off, 32);
    __shared__ float red[8];
    const int lane = threadIdx.x & 31, wave = threadIdx.x >> 5;
    if (lane == 0) red[wave] = s;
    __syncthreads();
    if (threadIdx.x == 0) {
        float t = 0.0f;
        #pragma unroll
        for (int i = 0; i < 8; ++i) t += red[i];
        float inv = 1.0f / fmaxf(sqrtf(t), 1e-8f);
        if (b < NWAY) pinv[b] = inv; else qinv[b - NWAY] = inv;
    }
}

// ---------------------------------------------------------------------------
// Kernel 4: cosine similarities sim[q][n] = <xq,proto_n> * qinv[q] * pinv[n]
//           (operands are L2-resident; 49 MFLOP -> scalar path is fine)
// ---------------------------------------------------------------------------
__global__ void cos_sim(const float* __restrict__ C, const float* __restrict__ proto,
                        const float* __restrict__ qinv, const float* __restrict__ pinv,
                        float* __restrict__ sim)
{
    int idx = blockIdx.x * blockDim.x + threadIdx.x;
    if (idx >= MQRY * NWAY) return;
    int q = idx / NWAY, n = idx - q * NWAY;
    const float* xq = C + (size_t)(MSUP + q) * DIM;
    const float* pn = proto + (size_t)n * DIM;
    float dot = 0.0f;
    for (int d = 0; d < DIM; ++d) dot = fmaf(xq[d], pn[d], dot);
    sim[idx] = dot * qinv[q] * pinv[n];
}

// ---------------------------------------------------------------------------
// Kernel 5: log_softmax over 32 classes, one wave32 per query row
// ---------------------------------------------------------------------------
__global__ void log_softmax32(const float* __restrict__ sim, float* __restrict__ out)
{
    const int q = blockIdx.x, lane = threadIdx.x;
    float v = sim[q * NWAY + lane];
    float m = v;
    #pragma unroll
    for (int off = 16; off >= 1; off >>= 1) m = fmaxf(m, __shfl_xor(m, off, 32));
    float e = __expf(v - m), s = e;
    #pragma unroll
    for (int off = 16; off >= 1; off >>= 1) s += __shfl_xor(s, off, 32);
    out[q * NWAY + lane] = v - m - __logf(s);
}

// ---------------------------------------------------------------------------
extern "C" void kernel_launch(void* const* d_in, const int* in_sizes, int n_in,
                              void* d_out, int out_size, void* d_ws, size_t ws_size,
                              hipStream_t stream)
{
    const float* supp  = (const float*)d_in[0];   // 32*5*3*84*84
    const float* query = (const float*)d_in[1];   // 32*15*3*84*84
    const float* Wenc  = (const float*)d_in[2];   // 21168*1600
    float* out = (float*)d_out;                   // 480*32

    float* C     = (float*)d_ws;                  // 640*1600
    float* proto = C     + (size_t)MROWS * DIM;   // 32*1600
    float* pinv  = proto + (size_t)NWAY  * DIM;   // 32
    float* qinv  = pinv  + NWAY;                  // 480
    float* sim   = qinv  + MQRY;                  // 480*32

    hipMemsetAsync(C, 0, (size_t)MROWS * DIM * sizeof(float), stream);

    dim3 grid(DIM / BN, MROWS / BM, SPLITK);      // 25 x 5 x 8 = 1000 WGs
    gemm_bf16_wmma<<<grid, BLK, 0, stream>>>(supp, query, Wenc, C);

    proto_mean<<<(NWAY * DIM + 255) / 256, 256, 0, stream>>>(C, proto);
    row_norms<<<NWAY + MQRY, 256, 0, stream>>>(C, proto, pinv, qinv);
    cos_sim<<<(MQRY * NWAY + 255) / 256, 256, 0, stream>>>(C, proto, qinv, pinv, sim);
    log_softmax32<<<MQRY, 32, 0, stream>>>(sim, out);
}
